// LearnableShapelet_75479755260552
// MI455X (gfx1250) — matrix operand order repaired and checked by reading, not compiled, not run on hardware
//
#include <hip/hip_runtime.h>

// ---------------------------------------------------------------------------
// Problem constants (from reference): x (64,8,4096) f32, shapelets (256,8,64)
// ---------------------------------------------------------------------------
constexpr int Bn = 64;          // batches
constexpr int Mc = 8;           // channels
constexpr int Sn = 4096;        // sequence length
constexpr int Kn = 256;         // shapelets
constexpr int Ln = 64;          // shapelet length
constexpr int Tn = Sn - Ln + 1; // 4033 windows
constexpr int Rn = Mc * Ln;     // 512 reduction dim

// Workspace layout (bytes)
constexpr size_t SZ_X    = (size_t)Bn * Mc * Sn * 2;   // bf16 x copy
constexpr size_t SZ_W    = (size_t)Rn * Kn * 2;        // bf16 W [r][k]
constexpr size_t OFF_XHI = 0;
constexpr size_t OFF_XLO = OFF_XHI + SZ_X;
constexpr size_t OFF_WHI = OFF_XLO + SZ_X;
constexpr size_t OFF_WLO = OFF_WHI + SZ_W;
constexpr size_t OFF_SSQ = OFF_WLO + SZ_W;             // K floats
constexpr size_t OFF_WSQ = OFF_SSQ + 1024;             // B*Sn floats (padded)
constexpr size_t OFF_MIN = OFF_WSQ + (size_t)Bn * Sn * 4; // B*K uint32

typedef __attribute__((ext_vector_type(16))) __bf16 v16bf;
typedef __attribute__((ext_vector_type(8)))  __bf16 bf16x8;
typedef __attribute__((ext_vector_type(8)))  float  v8f;
typedef __attribute__((ext_vector_type(4)))  float  v4f;
// reduced-alignment typedef: sliding windows make A-fragment loads 2B-aligned
typedef bf16x8 bf16x8u __attribute__((aligned(2)));

// ---------------------------------------------------------------------------
// order-preserving float <-> uint mapping for atomic min
// ---------------------------------------------------------------------------
__device__ inline unsigned f2ord(float f) {
  unsigned u = __float_as_uint(f);
  return (u & 0x80000000u) ? ~u : (u | 0x80000000u);
}
__device__ inline float ord2f(unsigned u) {
  unsigned v = (u & 0x80000000u) ? (u & 0x7FFFFFFFu) : ~u;
  return __uint_as_float(v);
}

// ---------------------------------------------------------------------------
// Prep kernels
// ---------------------------------------------------------------------------
__global__ void split_x_kernel(const float* __restrict__ x,
                               __bf16* __restrict__ xhi,
                               __bf16* __restrict__ xlo, int n) {
  int i = blockIdx.x * blockDim.x + threadIdx.x;
  if (i < n) {
    float f = x[i];
    __bf16 h = (__bf16)f;
    xhi[i] = h;
    xlo[i] = (__bf16)(f - (float)h);
  }
}

// shapelets (K, M, L) -> W[r][k] with r = m*L + l (bf16 hi/lo split)
__global__ void prep_w_kernel(const float* __restrict__ shp,
                              __bf16* __restrict__ whi,
                              __bf16* __restrict__ wlo) {
  int i = blockIdx.x * blockDim.x + threadIdx.x;
  if (i < Kn * Rn) {
    int k = i / Rn;
    int r = i % Rn;
    float f = shp[i];                 // shp[k*Rn + r], (M,L) contiguous per k
    __bf16 h = (__bf16)f;
    whi[r * Kn + k] = h;
    wlo[r * Kn + k] = (__bf16)(f - (float)h);
  }
}

__global__ void shpsq_kernel(const float* __restrict__ shp,
                             float* __restrict__ ssq) {
  int k = threadIdx.x;                // 256 threads, one block
  float s = 0.0f;
  for (int r = 0; r < Rn; ++r) {
    float v = shp[k * Rn + r];
    s += v * v;
  }
  ssq[k] = s;
}

// sliding-window sum of per-channel squares; pad [Tn, Sn) with +INF so the
// main kernel's min-reduction needs no bounds branches at all.
__global__ void winsq_kernel(const float* __restrict__ x,
                             float* __restrict__ winsq) {
  __shared__ float xsq[Sn];
  int b = blockIdx.x;
  const float* xb = x + (size_t)b * Mc * Sn;
  for (int s = threadIdx.x; s < Sn; s += blockDim.x) {
    float acc = 0.0f;
    for (int m = 0; m < Mc; ++m) {
      float v = xb[m * Sn + s];
      acc += v * v;
    }
    xsq[s] = acc;
  }
  __syncthreads();
  for (int t = threadIdx.x; t < Sn; t += blockDim.x) {
    if (t < Tn) {
      float acc = 0.0f;
      for (int l = 0; l < Ln; ++l) acc += xsq[t + l];
      winsq[b * Sn + t] = acc;
    } else {
      winsq[b * Sn + t] = __builtin_inff();
    }
  }
}

__global__ void init_min_kernel(unsigned* __restrict__ mb, int n) {
  int i = blockIdx.x * blockDim.x + threadIdx.x;
  if (i < n) mb[i] = 0xFFFFFFFFu;
}

__global__ void finalize_kernel(const unsigned* __restrict__ mb,
                                const float* __restrict__ ssq,
                                float* __restrict__ out, int n) {
  int i = blockIdx.x * blockDim.x + threadIdx.x;
  if (i < n) out[i] = ord2f(mb[i]) + ssq[i & (Kn - 1)];
}

// ---------------------------------------------------------------------------
// A-fragment load: 16x32 bf16, lane = (row = lane&15, half = lane>>4)
// elements 0..7  = r-offsets half*8 + 0..7        (8 contiguous bf16)
// elements 8..15 = r-offsets 16 + half*8 + 0..7   (8 contiguous bf16)
// ---------------------------------------------------------------------------
__device__ inline v16bf load_afrag(const __bf16* __restrict__ base,
                                   int t_row, int ll0, int half) {
  int e0 = t_row + ll0 + half * 8;
  bf16x8 g0 = *reinterpret_cast<const bf16x8u*>(base + e0);
  bf16x8 g1 = *reinterpret_cast<const bf16x8u*>(base + e0 + 16);
  v16bf a;
#pragma unroll
  for (int j = 0; j < 8; ++j) {
    a[j]     = g0[j];
    a[j + 8] = g1[j];
  }
  return a;
}

// ---------------------------------------------------------------------------
// Main kernel.
// Grid: (ceil(Tn/64)=64 t-blocks, 64 batches); block = 256 threads = 8 waves.
// Wave w -> tHalf = w>>2 (two 32-t halves of the 64-t block),
//           kGroup = w&3 (4 wave-groups cover K=256 with 64 k each).
// Each wave: 32t x 64k tile = c[2][4] fp32 fragments; reduction over R=512 in
// steps of 32 with 3 bf16 WMMAs per fragment per step (hi*hi + hi*lo + lo*hi):
// 24 v_wmma per step against 8 16-byte A loads and 128 2-byte B loads.
// ---------------------------------------------------------------------------
__global__ void __launch_bounds__(256) corr_min_kernel(
    const __bf16* __restrict__ xhi, const __bf16* __restrict__ xlo,
    const __bf16* __restrict__ whi, const __bf16* __restrict__ wlo,
    const float* __restrict__ winsq, unsigned* __restrict__ minbuf) {
  const int b    = blockIdx.y;
  const int wave = threadIdx.x >> 5;
  const int lane = threadIdx.x & 31;
  const int t0   = blockIdx.x * 64 + (wave >> 2) * 32;
  const int kb   = (wave & 3) * 64;
  const int half = lane >> 4;
  const int row  = lane & 15;

  v8f c[2][4] = {};                 // fp32 accumulators, [t-subtile][k-subtile]

  const __bf16* xb_hi = xhi + (size_t)b * Mc * Sn;
  const __bf16* xb_lo = xlo + (size_t)b * Mc * Sn;

  for (int r0 = 0; r0 < Rn; r0 += 32) {
    const int m   = r0 >> 6;        // channel
    const int ll0 = r0 & 63;        // 0 or 32 within the channel's 64 taps
    const __bf16* xmh = xb_hi + m * Sn;
    const __bf16* xml = xb_lo + m * Sn;

    v16bf ah[2], al[2];
#pragma unroll
    for (int ti = 0; ti < 2; ++ti) {
      int t_row = t0 + ti * 16 + row;
      if (t_row > Tn - 1) t_row = Tn - 1;   // clamp loads; INF-masked at min
      ah[ti] = load_afrag(xmh, t_row, ll0, half);
      al[ti] = load_afrag(xml, t_row, ll0, half);
    }

    // B fragment: lane l holds W[r0 + k][kcol = l&15], k in table order (8h+s / 16+8h+s-8)
    v16bf bh[4], bl[4];
#pragma unroll
    for (int kj = 0; kj < 4; ++kj) {
#pragma unroll
      for (int s = 0; s < 16; ++s) {
        const int rr = r0 + (s < 8 ? half * 8 + s : 16 + half * 8 + (s - 8));
        bh[kj][s] = whi[rr * Kn + kb + kj * 16 + row];
        bl[kj][s] = wlo[rr * Kn + kb + kj * 16 + row];
      }
    }

#pragma unroll
    for (int ti = 0; ti < 2; ++ti) {
#pragma unroll
      for (int kj = 0; kj < 4; ++kj) {
        c[ti][kj] = __builtin_amdgcn_wmma_f32_16x16x32_bf16(
            false, ah[ti], false, bh[kj], (short)0, c[ti][kj], false, false);
        c[ti][kj] = __builtin_amdgcn_wmma_f32_16x16x32_bf16(
            false, ah[ti], false, bl[kj], (short)0, c[ti][kj], false, false);
        c[ti][kj] = __builtin_amdgcn_wmma_f32_16x16x32_bf16(
            false, al[ti], false, bh[kj], (short)0, c[ti][kj], false, false);
      }
    }
  }

  // Branchless epilogue: dist'(t,k) = winsq[t] - 2*corr[t,k].
  // winsq is +INF-padded for t in [Tn, 4096), so clamped/garbage rows vanish
  // in the min automatically. D layout: element v of lane -> t-row =
  // ti*16 + v + 8*half, k-col = lane&15. t_base is a multiple of 8 -> aligned
  // v4f loads.
  const float* wsb = winsq + (size_t)b * Sn;
#pragma unroll
  for (int kj = 0; kj < 4; ++kj) {
    float best = __builtin_inff();
#pragma unroll
    for (int ti = 0; ti < 2; ++ti) {
      int t_base = t0 + ti * 16 + half * 8;
      v4f w0 = *reinterpret_cast<const v4f*>(wsb + t_base);
      v4f w1 = *reinterpret_cast<const v4f*>(wsb + t_base + 4);
#pragma unroll
      for (int v = 0; v < 4; ++v) {
        best = fminf(best, fmaf(-2.0f, c[ti][kj][v], w0[v]));
        best = fminf(best, fmaf(-2.0f, c[ti][kj][v + 4], w1[v]));
      }
    }
    best = fminf(best, __shfl_xor(best, 16, 32));   // fold the two row-halves
    if (half == 0) {
      atomicMin(minbuf + (size_t)b * Kn + kb + kj * 16 + row, f2ord(best));
    }
  }
}

// ---------------------------------------------------------------------------
extern "C" void kernel_launch(void* const* d_in, const int* in_sizes, int n_in,
                              void* d_out, int out_size, void* d_ws,
                              size_t ws_size, hipStream_t stream) {
  const float* x   = (const float*)d_in[0];
  const float* shp = (const float*)d_in[1];
  float* out       = (float*)d_out;
  char* ws         = (char*)d_ws;

  __bf16*   xhi = (__bf16*)(ws + OFF_XHI);
  __bf16*   xlo = (__bf16*)(ws + OFF_XLO);
  __bf16*   whi = (__bf16*)(ws + OFF_WHI);
  __bf16*   wlo = (__bf16*)(ws + OFF_WLO);
  float*    ssq = (float*)(ws + OFF_SSQ);
  float*    wsq = (float*)(ws + OFF_WSQ);
  unsigned* mnb = (unsigned*)(ws + OFF_MIN);

  const int nx = Bn * Mc * Sn;
  split_x_kernel<<<(nx + 255) / 256, 256, 0, stream>>>(x, xhi, xlo, nx);
  prep_w_kernel<<<(Kn * Rn + 255) / 256, 256, 0, stream>>>(shp, whi, wlo);
  shpsq_kernel<<<1, Kn, 0, stream>>>(shp, ssq);
  winsq_kernel<<<Bn, 256, 0, stream>>>(x, wsq);
  init_min_kernel<<<(Bn * Kn + 255) / 256, 256, 0, stream>>>(mnb, Bn * Kn);

  dim3 grid((Tn + 63) / 64, Bn);   // 64 x 64
  corr_min_kernel<<<grid, 256, 0, stream>>>(xhi, xlo, whi, wlo, wsq, mnb);

  finalize_kernel<<<(Bn * Kn + 255) / 256, 256, 0, stream>>>(mnb, ssq, out,
                                                             Bn * Kn);
}
